// BRNN_75857712382252
// MI455X (gfx1250) — compile-verified, matrix operand order checked by reading, not run on hardware
//
#include <hip/hip_runtime.h>
#include <math.h>

// Problem dims (fixed by the reference)
#define T_  512
#define B_  64
#define DX_ 128
#define H_  256
#define DZ_ 64

typedef __bf16          v16bf __attribute__((ext_vector_type(16)));
typedef float           v8f   __attribute__((ext_vector_type(8)));
typedef unsigned short  v16u  __attribute__((ext_vector_type(16)));
typedef unsigned short  v8u   __attribute__((ext_vector_type(8)));

union FragU { v16u u; v16bf b; };

// Software RNE f32->bf16 (cold prep path only; hot paths use native casts).
__device__ __forceinline__ unsigned short f2bf(float f) {
  union { float f; unsigned int u; } c; c.f = f;
  unsigned int u = c.u;
  u += 0x7FFFu + ((u >> 16) & 1u);
  return (unsigned short)(u >> 16);
}

// K index inside a 16x16x32 bf16 A/B fragment: per ISA 7.12.2 each lane's 16
// values are two contiguous 8-element runs: {0..7,16..23} (+8 for lanes>=16).
__device__ __forceinline__ int kOf(int j, int hi) {
  return ((j < 8) ? j : (j + 8)) + (hi ? 8 : 0);
}

// Pre-swizzled B fragment: frags[(kc*NT+nt)*32*16 + lane*16 + j]; 32B/lane,
// loads as 2x global_load_b128.
__device__ __forceinline__ v16bf load_b_frag(const unsigned short* frags, int NT,
                                             int kc, int nt, int lane) {
  const unsigned short* p = frags + ((((kc * NT + nt) << 5) + lane) << 4);
  FragU f;
  f.u = *(const v16u*)p;
  return f.b;
}

// A fragment from a row-major bf16 matrix in LDS: 2x ds_load_b128 per lane.
__device__ __forceinline__ v16bf load_a_lds(const __bf16* hb, int ldk,
                                            int m0, int kc, int lane) {
  const __bf16* p = hb + (m0 + (lane & 15)) * ldk + kc * 32 + ((lane >> 4) << 3);
  v8u a0 = *(const v8u*)p;
  v8u a1 = *(const v8u*)(p + 16);
  FragU f;
#pragma unroll
  for (int j = 0; j < 8; ++j) { f.u[j] = a0[j]; f.u[j + 8] = a1[j]; }
  return f.b;
}

__device__ __forceinline__ v8f wmma_bf16(v16bf a, v16bf b, v8f c) {
  return __builtin_amdgcn_wmma_f32_16x16x32_bf16(false, a, false, b, (short)0, c,
                                                 false, false);
}

// ---------------------------------------------------------------------------
// Kernel 0: swizzle W[N][K] (used as x @ W^T, so B[k][n] = W[n][k]) into the
// bf16 WMMA B-fragment layout.
// ---------------------------------------------------------------------------
__global__ void k_prep(const float* __restrict__ W, unsigned short* __restrict__ out,
                       int K, int N) {
  int idx = blockIdx.x * blockDim.x + threadIdx.x;
  if (idx >= K * N) return;
  int j    = idx & 15;
  int lane = (idx >> 4) & 31;
  int fi   = idx >> 9;
  int NT   = N >> 4;
  int nt = fi % NT, kc = fi / NT;
  int n = nt * 16 + (lane & 15);
  int k = kc * 32 + kOf(j, lane >= 16);
  out[idx] = f2bf(W[n * K + k]);
}

// ---------------------------------------------------------------------------
// Kernel 1: xproj[dir] = X(dir-ordered) @ Wih^T + (bih + bhh)
// Output in C-fragment order: xp[tile][lane][r], tile = (m0/16)*16 + nt.
// grid=(256,2), block=256 (8 waves); wave = one 16-row M tile x 16 N tiles.
// ---------------------------------------------------------------------------
__global__ __launch_bounds__(256) void k_xproj(
    const float* __restrict__ X,
    const unsigned short* __restrict__ wfF, const unsigned short* __restrict__ wfB,
    const float* __restrict__ bihF, const float* __restrict__ bhhF,
    const float* __restrict__ bihB, const float* __restrict__ bhhB,
    float* __restrict__ xpF, float* __restrict__ xpB)
{
  const int dir = blockIdx.y;
  const unsigned short* wf = dir ? wfB : wfF;
  const float* bih = dir ? bihB : bihF;
  const float* bhh = dir ? bhhB : bhhF;
  float* xp = dir ? xpB : xpF;

  const int wave = threadIdx.x >> 5;
  const int lane = threadIdx.x & 31;
  const int hi   = lane >> 4;
  const int m0   = blockIdx.x * 128 + wave * 16;

  // Per-lane A row (time-reversed indexing for the backward direction).
  int row = m0 + (lane & 15);
  int t = row >> 6, b = row & 63;
  const float* arow = X + (size_t)(dir ? ((T_ - 1 - t) * B_ + b) : row) * DX_;

  v8f acc[16];
#pragma unroll
  for (int q = 0; q < 16; ++q) acc[q] = {};

#pragma unroll
  for (int kc = 0; kc < 4; ++kc) {
    // Contiguous f32 runs -> vector loads + native bf16 converts.
    const float* p = arow + kc * 32 + (hi << 3);
    FragU fa;
#pragma unroll
    for (int j = 0; j < 8; ++j) {
      fa.b[j]     = (__bf16)p[j];
      fa.b[j + 8] = (__bf16)p[j + 16];
    }
#pragma unroll
    for (int nt = 0; nt < 16; ++nt)
      acc[nt] = wmma_bf16(fa.b, load_b_frag(wf, 16, kc, nt, lane), acc[nt]);
  }

#pragma unroll
  for (int nt = 0; nt < 16; ++nt) {
    int n = nt * 16 + (lane & 15);
    float bias = bih[n] + bhh[n];
    v8f o;
#pragma unroll
    for (int r = 0; r < 8; ++r) o[r] = acc[nt][r] + bias;
    size_t tile = (size_t)(m0 >> 4) * 16 + nt;
    *(v8f*)(xp + tile * 256 + lane * 8) = o;   // one 32B coalesced store
  }
}

// ---------------------------------------------------------------------------
// Kernel 2: recurrent scan, one persistent workgroup per direction (grid=2).
// h lives in LDS (bf16, row-major); per step h = tanh(xproj[t] + h @ Whh^T).
// Wave w: M-tile = w>>1, N-tiles (w&1)*8 .. +8  ->  64 WMMA/step/wave.
// hL/hR written in C-fragment order for the latent kernel.
// ---------------------------------------------------------------------------
__global__ __launch_bounds__(256) void k_scan(
    const unsigned short* __restrict__ whhF, const unsigned short* __restrict__ whhB,
    const float* __restrict__ xpF, const float* __restrict__ xpB,
    float* __restrict__ hL, float* __restrict__ hR)
{
  __shared__ __bf16 hbf[B_ * H_];   // 32 KB state mirror
  const int dir = blockIdx.x;
  const unsigned short* whh = dir ? whhB : whhF;
  const float* xp = dir ? xpB : xpF;
  float* hout = dir ? hR : hL;
  const int wave = threadIdx.x >> 5, lane = threadIdx.x & 31;
  const int mt = wave >> 1, ntBase = (wave & 1) * 8;
  const int hi = lane >> 4;

  for (int i = threadIdx.x; i < B_ * H_; i += 256) hbf[i] = (__bf16)0.0f;  // h0 = 0
  __syncthreads();

  for (int t = 0; t < T_; ++t) {
    v16bf a[8];
#pragma unroll
    for (int kc = 0; kc < 8; ++kc) a[kc] = load_a_lds(hbf, H_, mt * 16, kc, lane);

    v8f acc[8];
#pragma unroll
    for (int q = 0; q < 8; ++q) acc[q] = {};

#pragma unroll
    for (int kc = 0; kc < 8; ++kc)
#pragma unroll
      for (int q = 0; q < 8; ++q)
        acc[q] = wmma_bf16(a[kc], load_b_frag(whh, 16, kc, ntBase + q, lane), acc[q]);

    // Prefetch next step's xproj slab (64KB) into cache: 256 threads x 256B.
    if (t + 1 < T_) {
      const float* nxt = xp + (size_t)(t + 1) * (B_ * H_) + threadIdx.x * 64;
      __builtin_prefetch(nxt, 0, 1);
      __builtin_prefetch(nxt + 32, 0, 1);
    }

    __syncthreads();  // all reads of hbf complete before overwrite

    const int t_out = dir ? (T_ - 1 - t) : t;
#pragma unroll
    for (int q = 0; q < 8; ++q) {
      int nt = ntBase + q;
      int n  = nt * 16 + (lane & 15);
      size_t tile = (size_t)t * 64 + mt * 16 + nt;
      v8f xv = *(const v8f*)(xp + tile * 256 + lane * 8);   // 32B coalesced
      v8f hv;
#pragma unroll
      for (int r = 0; r < 8; ++r) {
        float v = tanhf(acc[q][r] + xv[r]);
        hv[r] = v;
        hbf[(mt * 16 + (hi << 3) + r) * H_ + n] = (__bf16)v;
      }
      size_t otile = (size_t)t_out * 64 + mt * 16 + nt;
      *(v8f*)(hout + otile * 256 + lane * 8) = hv;          // 32B coalesced
    }
    __syncthreads();
  }
}

// ---------------------------------------------------------------------------
// Kernel 3: latent scan (single persistent workgroup).
// Phase A: h = (tanh(z@Wt^T + bt) + hl + hr)/3    (M=64, N=256, K=64)
// Phase B: mu = h@Wmu^T + bmu ; std = softplus(h@Wsig^T + bsig); z = mu+std*eps
// ---------------------------------------------------------------------------
__global__ __launch_bounds__(256) void k_latent(
    const unsigned short* __restrict__ fWt, const unsigned short* __restrict__ fWmu,
    const unsigned short* __restrict__ fWsig,
    const float* __restrict__ bt, const float* __restrict__ bmu,
    const float* __restrict__ bsig,
    const float* __restrict__ hL, const float* __restrict__ hR,
    const float* __restrict__ eps, float* __restrict__ out)
{
  __shared__ __bf16 zbf[B_ * DZ_];  // 8 KB
  __shared__ __bf16 hbf[B_ * H_];   // 32 KB
  const int wave = threadIdx.x >> 5, lane = threadIdx.x & 31;
  const int hi = lane >> 4;
  const int mtA = wave >> 1, ntA = (wave & 1) * 8;   // phase A: N=256 -> 16 tiles
  const int mtB = wave & 3,  ntB = (wave >> 2) * 2;  // phase B: N=64  -> 4 tiles

  for (int i = threadIdx.x; i < B_ * DZ_; i += 256) zbf[i] = (__bf16)0.0f; // z0=0
  __syncthreads();

  const size_t slab = (size_t)T_ * B_ * DZ_;

  for (int t = 0; t < T_; ++t) {
    // -------- phase A --------
    v16bf az[2];
#pragma unroll
    for (int kc = 0; kc < 2; ++kc) az[kc] = load_a_lds(zbf, DZ_, mtA * 16, kc, lane);

#pragma unroll
    for (int q = 0; q < 8; ++q) {
      int nt = ntA + q;
      v8f acc = {};
#pragma unroll
      for (int kc = 0; kc < 2; ++kc)
        acc = wmma_bf16(az[kc], load_b_frag(fWt, 16, kc, nt, lane), acc);
      int n = nt * 16 + (lane & 15);
      float bb = bt[n];
      size_t tile = (size_t)t * 64 + mtA * 16 + nt;
      v8f lv = *(const v8f*)(hL + tile * 256 + lane * 8);   // 32B coalesced
      v8f rv = *(const v8f*)(hR + tile * 256 + lane * 8);
#pragma unroll
      for (int r = 0; r < 8; ++r) {
        int m = mtA * 16 + (hi << 3) + r;
        float v = (tanhf(acc[r] + bb) + lv[r] + rv[r]) * (1.0f / 3.0f);
        hbf[m * H_ + n] = (__bf16)v;
      }
    }
    __syncthreads();

    // -------- phase B --------
    v16bf ah[8];
#pragma unroll
    for (int kc = 0; kc < 8; ++kc) ah[kc] = load_a_lds(hbf, H_, mtB * 16, kc, lane);

#pragma unroll
    for (int p = 0; p < 2; ++p) {
      int nt = ntB + p;
      v8f am = {}, as = {};
#pragma unroll
      for (int kc = 0; kc < 8; ++kc) {
        am = wmma_bf16(ah[kc], load_b_frag(fWmu, 4, kc, nt, lane), am);
        as = wmma_bf16(ah[kc], load_b_frag(fWsig, 4, kc, nt, lane), as);
      }
      int n = nt * 16 + (lane & 15);
      float cbm = bmu[n], cbs = bsig[n];
#pragma unroll
      for (int r = 0; r < 8; ++r) {
        int m = mtB * 16 + (hi << 3) + r;
        float mu = am[r] + cbm;
        float sp = as[r] + cbs;
        float sd = (sp > 20.0f) ? sp : log1pf(__expf(sp));
        float e = eps[((size_t)t * B_ + m) * DZ_ + n];
        float z = mu + sd * e;
        size_t idx = (size_t)t * B_ * DZ_ + (size_t)m * DZ_ + n;
        out[idx]            = z;
        out[slab + idx]     = mu;
        out[2 * slab + idx] = sd;
        zbf[m * DZ_ + n] = (__bf16)z;
      }
    }
    __syncthreads();
  }
}

// ---------------------------------------------------------------------------
extern "C" void kernel_launch(void* const* d_in, const int* in_sizes, int n_in,
                              void* d_out, int out_size, void* d_ws, size_t ws_size,
                              hipStream_t stream) {
  (void)in_sizes; (void)n_in; (void)out_size; (void)ws_size;

  const float* X     = (const float*)d_in[0];
  const float* Wih_f = (const float*)d_in[1];
  const float* Whh_f = (const float*)d_in[2];
  const float* bih_f = (const float*)d_in[3];
  const float* bhh_f = (const float*)d_in[4];
  const float* Wih_b = (const float*)d_in[5];
  const float* Whh_b = (const float*)d_in[6];
  const float* bih_b = (const float*)d_in[7];
  const float* bhh_b = (const float*)d_in[8];
  const float* Wt    = (const float*)d_in[9];
  const float* bt    = (const float*)d_in[10];
  const float* Wmu   = (const float*)d_in[11];
  const float* bmu   = (const float*)d_in[12];
  const float* Wsig  = (const float*)d_in[13];
  const float* bsig  = (const float*)d_in[14];
  const float* eps   = (const float*)d_in[15];
  float* out = (float*)d_out;

  char* ws = (char*)d_ws;
  size_t off = 0;
  auto alloc = [&](size_t bytes) -> char* {
    char* p = ws + off;
    off += (bytes + 255) & ~(size_t)255;
    return p;
  };
  const size_t tbh = (size_t)T_ * B_ * H_ * sizeof(float);  // 33.5 MB each
  float* xpF = (float*)alloc(tbh);
  float* xpB = (float*)alloc(tbh);
  float* hL  = (float*)alloc(tbh);
  float* hR  = (float*)alloc(tbh);
  unsigned short* fWihF = (unsigned short*)alloc((size_t)DX_ * H_ * 2);
  unsigned short* fWihB = (unsigned short*)alloc((size_t)DX_ * H_ * 2);
  unsigned short* fWhhF = (unsigned short*)alloc((size_t)H_ * H_ * 2);
  unsigned short* fWhhB = (unsigned short*)alloc((size_t)H_ * H_ * 2);
  unsigned short* fWt   = (unsigned short*)alloc((size_t)DZ_ * H_ * 2);
  unsigned short* fWmu  = (unsigned short*)alloc((size_t)H_ * DZ_ * 2);
  unsigned short* fWsig = (unsigned short*)alloc((size_t)H_ * DZ_ * 2);

  auto prep = [&](const float* W, unsigned short* dst, int K, int N) {
    int total = K * N;
    k_prep<<<(total + 255) / 256, 256, 0, stream>>>(W, dst, K, N);
  };
  prep(Wih_f, fWihF, DX_, H_);
  prep(Wih_b, fWihB, DX_, H_);
  prep(Whh_f, fWhhF, H_,  H_);
  prep(Whh_b, fWhhB, H_,  H_);
  prep(Wt,    fWt,   DZ_, H_);
  prep(Wmu,   fWmu,  H_,  DZ_);
  prep(Wsig,  fWsig, H_,  DZ_);

  dim3 gx(T_ * B_ / 128, 2);
  k_xproj<<<gx, 256, 0, stream>>>(X, fWihF, fWihB, bih_f, bhh_f, bih_b, bhh_b,
                                  xpF, xpB);
  k_scan<<<2, 256, 0, stream>>>(fWhhF, fWhhB, xpF, xpB, hL, hR);
  k_latent<<<1, 256, 0, stream>>>(fWt, fWmu, fWsig, bt, bmu, bsig, hL, hR, eps, out);
}
